// MultiheadAttention_85950885528461
// MI455X (gfx1250) — matrix operand order repaired and checked by reading, not compiled, not run on hardware
//
#include <hip/hip_runtime.h>

// ---------------------------------------------------------------------------
// MultiheadAttention forward for gfx1250 (MI455X), wave32 + WMMA f16.
// T=2048, B=2, E=1024, H=16, hd=64, SCALE=1/8.
// ---------------------------------------------------------------------------

typedef __attribute__((ext_vector_type(16))) _Float16 v16h;
typedef __attribute__((ext_vector_type(8)))  _Float16 v8h;
typedef __attribute__((ext_vector_type(8)))  float    v8f;

#define T_LEN   2048
#define B_SZ    2
#define EMBED   1024
#define HEADS   16
#define HD      64
#define M_ROWS  (T_LEN * B_SZ)   // 4096
#define S_LEN   2048
#define BH      (B_SZ * HEADS)   // 32

__device__ __forceinline__ v8f wmma_f16(v16h a, v16h b, v8f c) {
  // 8 args: (neg_a, A, neg_b, B, c_mod, C, reuse_a, reuse_b)
  return __builtin_amdgcn_wmma_f32_16x16x32_f16(
      false, a, false, b, (short)0, c, false, false);
}

__device__ __forceinline__ v16h cat8(v8h lo, v8h hi) {
  return __builtin_shufflevector(lo, hi, 0, 1, 2, 3, 4, 5, 6, 7, 8, 9, 10, 11,
                                 12, 13, 14, 15);
}

// ---------------------------------------------------------------------------
// Kernel 1: fused projection  y = x @ W^T + bias  (x: [T,B,E] fp32 = [M,K])
// MODE 0: Q -> qh[bh][t][d] f16, scaled by 1/8
// MODE 1: K -> kh[bh][s][d] f16
// MODE 2: V -> vt[bh][d][s] f16 (transposed for contiguous PV B-fragments)
// grid (M/16, E/256), block 256; each wave owns TWO 16x16 N-tiles so the A
// fragment is reused by 2 WMMAs per k-step (halves A traffic per output).
// ---------------------------------------------------------------------------
template <int MODE>
__global__ __launch_bounds__(256) void proj_qkv_kernel(
    const float* __restrict__ x, const float* __restrict__ W,
    const float* __restrict__ bias, _Float16* __restrict__ out) {
  const int K = EMBED;
  const int m0   = blockIdx.x * 16;
  const int wave = threadIdx.x >> 5;
  const int lane = threadIdx.x & 31;
  const int row  = lane & 15;
  const int hi   = lane >> 4;          // 0 or 1
  const int n    = blockIdx.y * 256 + wave * 16 + row;   // B-frag column
  const int n2   = n + 128;

  const float* arow  = x + (size_t)(m0 + row) * K;
  const float* brow  = W + (size_t)n * K;
  const float* brow2 = W + (size_t)n2 * K;
  const int akb = hi ? 16 : 0;
  const int bkb = hi ? 16 : 0;

  v8f acc0 = {}, acc1 = {};
  for (int k = 0; k < K; k += 32) {
    __builtin_prefetch(brow + k + 128, 0, 3);    // global_prefetch_b8
    __builtin_prefetch(brow2 + k + 128, 0, 3);
    v16h a, b0, b1;
#pragma unroll
    for (int e = 0; e < 8; ++e) {
      a[e]     = (_Float16)arow[k + akb + e];
      a[e + 8] = (_Float16)arow[k + akb + 8 + e];
    }
#pragma unroll
    for (int e = 0; e < 16; ++e) {
      b0[e] = (_Float16)brow[k + bkb + e];
      b1[e] = (_Float16)brow2[k + bkb + e];
    }
    acc0 = wmma_f16(a, b0, acc0);
    acc1 = wmma_f16(a, b1, acc1);
  }

  auto store_tile = [&](int nn, v8f acc) {
    const float bs = bias[nn];
    const int h = nn >> 6, d = nn & 63;
#pragma unroll
    for (int r = 0; r < 8; ++r) {
      const int m  = m0 + r + hi * 8;
      const int t  = m >> 1;           // m = t*B + b, B=2
      const int bb = m & 1;
      const int bh = bb * HEADS + h;
      float v = acc[r] + bs;
      if (MODE == 0) v *= 0.125f;      // SCALE = 64^-0.5, folded at compile time
      const size_t idx = (MODE == 2)
                             ? ((size_t)bh * HD + d) * S_LEN + t   // vt[bh][d][s]
                             : ((size_t)bh * T_LEN + t) * HD + d;  // qh/kh[bh][t][d]
      out[idx] = (_Float16)v;
    }
  };
  store_tile(n, acc0);
  store_tile(n2, acc1);
}

// ---------------------------------------------------------------------------
// Kernel 2: attention core. One block = (batch b, 16 query rows), loops all
// 16 heads; scores + avg accumulate in LDS (2 x 128KB of the 320KB WGP LDS).
// grid (T/16, B), block 256 (8 waves), dynamic LDS = 267,328 B.
// ---------------------------------------------------------------------------
__global__ __launch_bounds__(256) void attn_kernel(
    const _Float16* __restrict__ qh, const _Float16* __restrict__ kh,
    const _Float16* __restrict__ vt, _Float16* __restrict__ attn16,
    float* __restrict__ avg_out) {
  extern __shared__ char smem_raw[];
  float* sc    = (float*)smem_raw;          // [16][2048] scores -> probs
  float* avg   = sc + 16 * S_LEN;           // [16][2048] head-avg probs
  float* red   = avg + 16 * S_LEN;          // [16][16] row partials
  float* rstat = red + 16 * 16;             // [16] row max / inv-sum
  float* red2  = rstat + 16;                // [4][16][16] partial O tiles

  const int t0   = blockIdx.x * 16;
  const int b    = blockIdx.y;
  const int tid  = threadIdx.x;
  const int wave = tid >> 5;
  const int lane = tid & 31;
  const int row  = lane & 15;
  const int hi   = lane >> 4;
  const int akb  = hi ? 16 : 0;
  const int bkb  = hi ? 16 : 0;

  for (int i = tid; i < 16 * S_LEN; i += 256) avg[i] = 0.0f;

  for (int h = 0; h < HEADS; ++h) {
    const int bh = b * HEADS + h;

    // --- Q fragments for this head (16 rows x 64 d = 2 K-chunks) ---
    const _Float16* qrow = qh + ((size_t)bh * T_LEN + t0 + row) * HD;
    const v16h aq0 = cat8(*(const v8h*)(qrow + akb),
                          *(const v8h*)(qrow + akb + 8));
    const v16h aq1 = cat8(*(const v8h*)(qrow + 32 + akb),
                          *(const v8h*)(qrow + 32 + akb + 8));

    // --- scores: S = Q @ K^T, 128 column tiles striped over 8 waves ---
    for (int ct = wave; ct < S_LEN / 16; ct += 8) {
      const int s_col = ct * 16 + row;
      const _Float16* krow = kh + ((size_t)bh * S_LEN + s_col) * HD;
      const v16h b0 = *(const v16h*)(krow + bkb);
      const v16h b1 = *(const v16h*)(krow + 32 + bkb);
      v8f c = {};
      c = wmma_f16(aq0, b0, c);
      c = wmma_f16(aq1, b1, c);
#pragma unroll
      for (int r = 0; r < 8; ++r)
        sc[(r + hi * 8) * S_LEN + ct * 16 + row] = c[r];
    }
    __syncthreads();

    // --- softmax over S per row: 16 threads per row ---
    {
      const int rr  = tid & 15;
      const int seg = tid >> 4;
      float mx = -1e30f;
      for (int s = seg; s < S_LEN; s += 16) mx = fmaxf(mx, sc[rr * S_LEN + s]);
      red[rr * 16 + seg] = mx;
      __syncthreads();
      if (tid < 16) {
        float m_ = red[tid * 16];
        for (int j = 1; j < 16; ++j) m_ = fmaxf(m_, red[tid * 16 + j]);
        rstat[tid] = m_;
      }
      __syncthreads();
      const float m_ = rstat[rr];
      float sum = 0.0f;
      for (int s = seg; s < S_LEN; s += 16) {
        const float p = __expf(sc[rr * S_LEN + s] - m_);
        sc[rr * S_LEN + s] = p;
        sum += p;
      }
      red[rr * 16 + seg] = sum;
      __syncthreads();
      if (tid < 16) {
        float l_ = 0.0f;
        for (int j = 0; j < 16; ++j) l_ += red[tid * 16 + j];
        rstat[tid] = 1.0f / l_;
      }
      __syncthreads();
      const float inv = rstat[rr];
      for (int s = seg; s < S_LEN; s += 16) {
        const float p = sc[rr * S_LEN + s] * inv;
        sc[rr * S_LEN + s] = p;
        avg[rr * S_LEN + s] += p * (1.0f / HEADS);
      }
    }
    __syncthreads();

    // --- O = P @ V : wave w -> d-tile (w&3), S-half (w>>2) ---
    {
      const int nd = wave & 3, sh = wave >> 2;
      const int d  = nd * 16 + row;
      const _Float16* vrow = vt + ((size_t)bh * HD + d) * S_LEN;
      v8f o = {};
      for (int kc = 0; kc < 32; ++kc) {
        const int s0 = sh * 1024 + kc * 32;
        // A (probs) fragment from LDS via b128 reads
        const float4* p0 = (const float4*)&sc[row * S_LEN + s0 + akb];
        const float4* p1 = (const float4*)&sc[row * S_LEN + s0 + akb + 8];
        const float4 f0 = p0[0], f1 = p0[1], f2 = p1[0], f3 = p1[1];
        v16h ap;
        ap[0] = (_Float16)f0.x;  ap[1] = (_Float16)f0.y;
        ap[2] = (_Float16)f0.z;  ap[3] = (_Float16)f0.w;
        ap[4] = (_Float16)f1.x;  ap[5] = (_Float16)f1.y;
        ap[6] = (_Float16)f1.z;  ap[7] = (_Float16)f1.w;
        ap[8] = (_Float16)f2.x;  ap[9] = (_Float16)f2.y;
        ap[10] = (_Float16)f2.z; ap[11] = (_Float16)f2.w;
        ap[12] = (_Float16)f3.x; ap[13] = (_Float16)f3.y;
        ap[14] = (_Float16)f3.z; ap[15] = (_Float16)f3.w;
        const v16h bv = *(const v16h*)(vrow + s0 + bkb);
        o = wmma_f16(ap, bv, o);
      }
      if (wave >= 4) {
#pragma unroll
        for (int r = 0; r < 8; ++r)
          red2[(nd * 16 + r + hi * 8) * 16 + row] = o[r];
      }
      __syncthreads();
      if (wave < 4) {
#pragma unroll
        for (int r = 0; r < 8; ++r) {
          const float v = o[r] + red2[(nd * 16 + r + hi * 8) * 16 + row];
          const int t = t0 + r + hi * 8;
          attn16[((size_t)t * B_SZ + b) * EMBED + h * HD + d] = (_Float16)v;
        }
      }
      __syncthreads();
    }
  }

  // --- write head-averaged weights: avg_out[b][t][s] (streaming, NT) ---
  for (int i = tid; i < 16 * S_LEN; i += 256) {
    const int rr = i >> 11, s = i & (S_LEN - 1);
    __builtin_nontemporal_store(
        avg[i], &avg_out[((size_t)b * T_LEN + t0 + rr) * S_LEN + s]);
  }
}

// ---------------------------------------------------------------------------
// Kernel 3: output projection  out = attn @ Wo^T + bo  (f16 A, fp32 NT out)
// grid (M/16, E/256), block 256; two N-tiles per wave.
// ---------------------------------------------------------------------------
__global__ __launch_bounds__(256) void proj_out_kernel(
    const _Float16* __restrict__ x, const float* __restrict__ W,
    const float* __restrict__ bias, float* __restrict__ out) {
  const int K = EMBED;
  const int m0   = blockIdx.x * 16;
  const int wave = threadIdx.x >> 5;
  const int lane = threadIdx.x & 31;
  const int row  = lane & 15;
  const int hi   = lane >> 4;
  const int n    = blockIdx.y * 256 + wave * 16 + row;
  const int n2   = n + 128;

  const _Float16* arow  = x + (size_t)(m0 + row) * K;
  const float*    brow  = W + (size_t)n * K;
  const float*    brow2 = W + (size_t)n2 * K;
  const int akb = hi ? 16 : 0;
  const int bkb = hi ? 16 : 0;

  v8f acc0 = {}, acc1 = {};
  for (int k = 0; k < K; k += 32) {
    __builtin_prefetch(brow + k + 128, 0, 3);
    __builtin_prefetch(brow2 + k + 128, 0, 3);
    const v16h a = cat8(*(const v8h*)(arow + k + akb),
                        *(const v8h*)(arow + k + akb + 8));
    v16h b0, b1;
#pragma unroll
    for (int e = 0; e < 16; ++e) {
      b0[e] = (_Float16)brow[k + bkb + e];
      b1[e] = (_Float16)brow2[k + bkb + e];
    }
    acc0 = wmma_f16(a, b0, acc0);
    acc1 = wmma_f16(a, b1, acc1);
  }

  const float bs0 = bias[n];
  const float bs1 = bias[n2];
#pragma unroll
  for (int r = 0; r < 8; ++r) {
    const size_t mrow = (size_t)(m0 + r + hi * 8) * EMBED;
    __builtin_nontemporal_store(acc0[r] + bs0, &out[mrow + n]);
    __builtin_nontemporal_store(acc1[r] + bs1, &out[mrow + n2]);
  }
}

// ---------------------------------------------------------------------------
extern "C" void kernel_launch(void* const* d_in, const int* in_sizes, int n_in,
                              void* d_out, int out_size, void* d_ws, size_t ws_size,
                              hipStream_t stream) {
  (void)in_sizes; (void)n_in; (void)out_size; (void)ws_size;
  const float* query = (const float*)d_in[0];
  const float* key   = (const float*)d_in[1];
  const float* value = (const float*)d_in[2];
  const float* Wq = (const float*)d_in[3];
  const float* bq = (const float*)d_in[4];
  const float* Wk = (const float*)d_in[5];
  const float* bk = (const float*)d_in[6];
  const float* Wv = (const float*)d_in[7];
  const float* bv = (const float*)d_in[8];
  const float* Wo = (const float*)d_in[9];
  const float* bo = (const float*)d_in[10];

  float* out = (float*)d_out;                          // [T,B,E]
  float* avg = out + (size_t)T_LEN * B_SZ * EMBED;     // [B,T,S]

  // workspace: 4 x 8MB f16 tensors = 32MB
  _Float16* qh     = (_Float16*)d_ws;                      // [BH][T][hd]
  _Float16* kh     = qh + (size_t)BH * T_LEN * HD;         // [BH][S][hd]
  _Float16* vt     = kh + (size_t)BH * S_LEN * HD;         // [BH][hd][S]
  _Float16* attn16 = vt + (size_t)BH * HD * S_LEN;         // [T,B,E]

  const dim3 blk(256);
  const dim3 gproj(M_ROWS / 16, EMBED / 256);
  proj_qkv_kernel<0><<<gproj, blk, 0, stream>>>(query, Wq, bq, qh);
  proj_qkv_kernel<1><<<gproj, blk, 0, stream>>>(key,   Wk, bk, kh);
  proj_qkv_kernel<2><<<gproj, blk, 0, stream>>>(value, Wv, bv, vt);

  const size_t smem =
      (size_t)(2 * 16 * S_LEN + 16 * 16 + 16 + 4 * 16 * 16) * sizeof(float);
  attn_kernel<<<dim3(T_LEN / 16, B_SZ), blk, smem, stream>>>(qh, kh, vt,
                                                             attn16, avg);

  proj_out_kernel<<<gproj, blk, 0, stream>>>(attn16, Wo, bo, out);
}
